// EGNN_dynamics_AD2_27006754357331
// MI455X (gfx1250) — matrix operand order repaired and checked by reading, not compiled, not run on hardware
//
// EGNN dynamics, fully fused per-graph kernel for gfx1250 (MI455X).
// One workgroup (128 thr = 4 waves) per graph; h/x/segment-sums LDS-resident
// across all 4 layers; all GEMMs via v_wmma_f32_16x16x32_bf16.
// Round 3: v_perm_b32-based bf16 pair packing (3 VALU) and hardware
// v_exp2/v_rcp silu to shrink the VALU tail around the WMMA hot loop.
#include <hip/hip_runtime.h>

typedef __attribute__((ext_vector_type(16))) __bf16 vbf16;
typedef __attribute__((ext_vector_type(8)))  float  v8f;

union Frag { vbf16 v; unsigned int u[8]; };

// ---- LDS partition (dword offsets) ----
#define O_W1    0        // edge_w1  swizzled bf16 pairs, 5120 dw (K padded 130->160)
#define O_W2    5120     // edge_w2  2048
#define O_CM1   7168
#define O_CC1   9216
#define O_NW1   11264    // node_w1  4096
#define O_NW2   15360    // 2048
#define O_EB1   17408
#define O_EB2   17472
#define O_NB1   17536
#define O_NB2   17600
#define O_CMB   17664
#define O_CCB   17728
#define O_CMW2  17792
#define O_CCW2  17856
#define O_H     17920    // h        [32][64] f32 master
#define O_SEG   19968    // seg(ef)  [32][64] f32 (atomic target)
#define O_HBF   22016    // bf16 mirror of h, pairs [32][36]
#define O_SEGBF 23168    // bf16 mirror of seg,    [32][36]
#define O_X     24320    // [32][4]
#define O_X0    24448
#define O_XA    24576
#define O_MEAN  24704    // (+pad)
#define O_ESCR  24708    // per-wave edge scalars [4][16][12]
#define O_SCRB  25476    // per-wave packed scratch [4][16][36] u32 pairs
#define SMEM_DW 27780    // ~111 KB -> 2 WGs per 320KB WGP

// Two f32 -> packed bf16x2: round-half-up then v_perm_b32 splices the high
// halves. 3 VALU ops; differs from RNE only on exact ties.
__device__ __forceinline__ unsigned int pk_bf16(float a, float b) {
  unsigned int ua = __builtin_bit_cast(unsigned int, a) + 0x8000u;
  unsigned int ub = __builtin_bit_cast(unsigned int, b) + 0x8000u;
  return __builtin_amdgcn_perm(ub, ua, 0x07060302u);
}

// silu via hardware transcendentals (v_exp_f32 / v_rcp_f32, ~1 ulp):
// x * rcp(1 + exp2(-x * log2(e)))
__device__ __forceinline__ float silu_f(float v) {
  return v * __builtin_amdgcn_rcpf(1.0f +
             __builtin_amdgcn_exp2f(v * -1.442695040888963f));
}

__device__ __forceinline__ v8f wmma_bf(Frag a, Frag b, v8f c) {
  return __builtin_amdgcn_wmma_f32_16x16x32_bf16(false, a.v, false, b.v,
                                                 (short)0, c, false, false);
}

// A fragments whose 8 dwords per lane are two 4-dword runs 8 dwords apart: two ds_load_b128.
__device__ __forceinline__ Frag frag_b128(const unsigned int* base) {
  Frag a;
  const uint4* p = (const uint4*)base;
  uint4 q0 = p[0], q1 = p[2];
  a.u[0] = q0.x; a.u[1] = q0.y; a.u[2] = q0.z; a.u[3] = q0.w;
  a.u[4] = q1.x; a.u[5] = q1.y; a.u[6] = q1.z; a.u[7] = q1.w;
  return a;
}

// A fragment from a pair-packed [row][36] image: row stride 144B (16B aligned,
// bank-spread); VGPR v holds K-pairs {g*4 + (v&3)} and {+8} within the k-tile.
__device__ __forceinline__ Frag afrag_pairs(const unsigned int* img, int row,
                                            int kt, int g) {
  return frag_b128(img + row * 36 + kt * 16 + g * 4);
}

// B fragment from the swizzled weight image: dwords contiguous per lane.
__device__ __forceinline__ Frag load_bfrag(const unsigned int* w, int kt, int nt,
                                           int lane) {
  Frag b;
  const uint4* p = (const uint4*)(w + ((kt * 4 + nt) * 32 + lane) * 8);
  uint4 q0 = p[0], q1 = p[1];
  b.u[0] = q0.x; b.u[1] = q0.y; b.u[2] = q0.z; b.u[3] = q0.w;
  b.u[4] = q1.x; b.u[5] = q1.y; b.u[6] = q1.z; b.u[7] = q1.w;
  return b;
}

// global f32 [Krows][64] -> LDS swizzled bf16-pair image (zero padded):
// dst[((kt*4+nt)*32 + lane)*8 + v] so each B fragment is lane-contiguous.
__device__ __forceinline__ void stage_w(unsigned int* dst, const float* src,
                                        int Krows, int k2cnt, int tid) {
  for (int i = tid; i < k2cnt * 64; i += 128) {
    int k2 = i >> 6, n = i & 63, k = k2 << 1;
    float a = (k     < Krows) ? src[k * 64 + n]      : 0.f;
    float b = (k + 1 < Krows) ? src[k * 64 + 64 + n] : 0.f;
    int kt = k2 >> 4, g = (k2 >> 2) & 1, v = (k2 & 3) | ((k2 >> 1) & 4);
    int nt = n >> 4, ln = n & 15;
    dst[(((kt * 4 + nt) * 2 + g) * 16 + ln) * 8 + v] = pk_bf16(a, b);
  }
}

// f32 [32][64] -> pair image [32][36]
__device__ __forceinline__ void pack_rows(unsigned int* dst, const float* src,
                                          int tid) {
  for (int i = tid; i < 1024; i += 128) {
    int r = i >> 5, p = i & 31;
    dst[r * 36 + p] = pk_bf16(src[r * 64 + 2 * p], src[r * 64 + 2 * p + 1]);
  }
}

__global__ __launch_bounds__(128) void egnn_fused_kernel(
    const float* __restrict__ inx,  const float* __restrict__ tarr,
    const float* __restrict__ embw, const float* __restrict__ embb,
    const float* __restrict__ ew1,  const float* __restrict__ eb1,
    const float* __restrict__ ew2,  const float* __restrict__ eb2,
    const float* __restrict__ nw1,  const float* __restrict__ nb1,
    const float* __restrict__ nw2,  const float* __restrict__ nb2,
    const float* __restrict__ cmw1, const float* __restrict__ cmb1,
    const float* __restrict__ cmw2,
    const float* __restrict__ ccw1, const float* __restrict__ ccb1,
    const float* __restrict__ ccw2,
    float* __restrict__ out) {
  extern __shared__ float smem[];
  const int tid  = threadIdx.x;
  const int wave = tid >> 5, lane = tid & 31;
  const int g = lane >> 4, ln = lane & 15;
  const int b = blockIdx.x;

  float* H     = smem + O_H;
  float* SEG   = smem + O_SEG;
  float* X     = smem + O_X;
  float* X0    = smem + O_X0;
  float* XA    = smem + O_XA;
  unsigned int* HBF   = (unsigned int*)(smem + O_HBF);
  unsigned int* SEGBF = (unsigned int*)(smem + O_SEGBF);
  unsigned int* W1  = (unsigned int*)(smem + O_W1);
  unsigned int* W2  = (unsigned int*)(smem + O_W2);
  unsigned int* CM1 = (unsigned int*)(smem + O_CM1);
  unsigned int* CC1 = (unsigned int*)(smem + O_CC1);
  unsigned int* NW1 = (unsigned int*)(smem + O_NW1);
  unsigned int* NW2 = (unsigned int*)(smem + O_NW2);
  float* escr  = smem + O_ESCR + wave * 192;                 // [16][12] per wave
  int*   escri = (int*)escr;
  unsigned int* scrbf = (unsigned int*)(smem + O_SCRB) + wave * 576;  // [16][36]

  // ---- init: coords, embedding h = onehot@emb_w + t*emb_w[22] + emb_b ----
  {
    float t = tarr[b];
    const float* xin = inx + b * 66;
    for (int i = tid; i < 66; i += 128) {
      int p = i / 3, d = i - p * 3;
      float v = xin[i];
      X[p * 4 + d] = v; X0[p * 4 + d] = v;
    }
    XA[tid] = 0.f;          // 128 slots, one per thread
    for (int i = tid; i < 2048; i += 128) {
      int p = i >> 6, k = i & 63;
      SEG[i] = 0.f;
      H[i] = (p < 22) ? (embw[p * 64 + k] + t * embw[22 * 64 + k] + embb[k]) : 0.f;
    }
  }
  __syncthreads();

  for (int l = 0; l < 4; ++l) {
    // ---- stage this layer's weights + refresh bf16 mirror of h ----
    stage_w(W1,  ew1  + l * 130 * 64, 130, 80, tid);
    stage_w(W2,  ew2  + l * 4096,      64, 32, tid);
    stage_w(CM1, cmw1 + l * 4096,      64, 32, tid);
    stage_w(CC1, ccw1 + l * 4096,      64, 32, tid);
    stage_w(NW1, nw1  + l * 8192,     128, 64, tid);
    stage_w(NW2, nw2  + l * 4096,      64, 32, tid);
    pack_rows(HBF, H, tid);
    for (int i = tid; i < 64; i += 128) {
      smem[O_EB1 + i]  = eb1[l * 64 + i];  smem[O_EB2 + i]  = eb2[l * 64 + i];
      smem[O_NB1 + i]  = nb1[l * 64 + i];  smem[O_NB2 + i]  = nb2[l * 64 + i];
      smem[O_CMB + i]  = cmb1[l * 64 + i]; smem[O_CCB + i]  = ccb1[l * 64 + i];
      smem[O_CMW2 + i] = cmw2[l * 64 + i]; smem[O_CCW2 + i] = ccw2[l * 64 + i];
    }
    __syncthreads();

    // ---- edge phase: 29 tiles of 16 edges, round-robin over 4 waves ----
    for (int tile = wave; tile < 29; tile += 4) {
      if (g == 0) {  // lanes 0..15: per-edge geometry
        int e = tile * 16 + ln;
        int r = 0, c = 0;
        float rad = 0.f, ea = 0.f;
        float cd0 = 0, cd1 = 0, cd2 = 0, cr0 = 0, cr1 = 0, cr2 = 0;
        if (e < 462) {
          r = e / 21; int q = e - r * 21; c = q + (q >= r);
          float xr0 = X[r * 4], xr1 = X[r * 4 + 1], xr2 = X[r * 4 + 2];
          float xc0 = X[c * 4], xc1 = X[c * 4 + 1], xc2 = X[c * 4 + 2];
          cd0 = xr0 - xc0; cd1 = xr1 - xc1; cd2 = xr2 - xc2;
          rad = cd0 * cd0 + cd1 * cd1 + cd2 * cd2;
          float inv = 1.f / (sqrtf(rad + 1e-8f) + 1.f);
          cd0 *= inv; cd1 *= inv; cd2 *= inv;
          float d0 = X0[r * 4] - X0[c * 4], d1 = X0[r * 4 + 1] - X0[c * 4 + 1],
                d2 = X0[r * 4 + 2] - X0[c * 4 + 2];
          ea = d0 * d0 + d1 * d1 + d2 * d2;      // edge_attr (fixed coords)
          cr0 = xr1 * xc2 - xr2 * xc1;
          cr1 = xr2 * xc0 - xr0 * xc2;
          cr2 = xr0 * xc1 - xr1 * xc0;
          float cn = cr0 * cr0 + cr1 * cr1 + cr2 * cr2;
          float ic = 1.f / (sqrtf(cn + 1e-8f) + 1.f);
          cr0 *= ic; cr1 *= ic; cr2 *= ic;
        }
        escr[ln * 12 + 0] = rad; escr[ln * 12 + 1] = ea;
        escr[ln * 12 + 2] = cd0; escr[ln * 12 + 3] = cd1; escr[ln * 12 + 4] = cd2;
        escr[ln * 12 + 5] = cr0; escr[ln * 12 + 6] = cr1; escr[ln * 12 + 7] = cr2;
        escri[ln * 12 + 10] = r; escri[ln * 12 + 11] = c;
      }
      __builtin_amdgcn_wave_barrier();

      int er_ = escri[ln * 12 + 10], ec_ = escri[ln * 12 + 11];
      float radm = escr[ln * 12 + 0], eam = escr[ln * 12 + 1];

      // GEMM1: [16 x 160] e_in @ edge_w1 -> silu  (A frags: pure b128 loads)
      v8f acc[4] = {};
#pragma unroll
      for (int kt = 0; kt < 5; ++kt) {
        Frag a;
        if (kt < 2)      a = afrag_pairs(HBF, er_, kt, g);
        else if (kt < 4) a = afrag_pairs(HBF, ec_, kt - 2, g);
        else {
          unsigned int re = pk_bf16(radm, eam);
#pragma unroll
          for (int v = 0; v < 8; ++v) a.u[v] = 0u;
          a.u[0] = (g == 0) ? re : 0u;
        }
#pragma unroll
        for (int nt = 0; nt < 4; ++nt)
          acc[nt] = wmma_bf(a, load_bfrag(W1, kt, nt, lane), acc[nt]);
      }
      // silu(+b1) -> packed scratch (pair partner via lane shuffle)
#pragma unroll
      for (int nt = 0; nt < 4; ++nt) {
        float bia = smem[O_EB1 + nt * 16 + ln];
#pragma unroll
        for (int j = 0; j < 8; ++j) {
          float v  = silu_f(acc[nt][j] + bia);
          float vo = __shfl_xor(v, 1);
          if (!(ln & 1))
            scrbf[(g * 8 + j) * 36 + nt * 8 + (ln >> 1)] = pk_bf16(v, vo);
        }
      }
      __builtin_amdgcn_wave_barrier();

      // GEMM2: hidden @ edge_w2 -> silu = ef ; fused seg(ef) atomics
      v8f ef[4] = {};
#pragma unroll
      for (int kt = 0; kt < 2; ++kt) {
        Frag a = afrag_pairs(scrbf, ln, kt, g);
#pragma unroll
        for (int nt = 0; nt < 4; ++nt)
          ef[nt] = wmma_bf(a, load_bfrag(W2, kt, nt, lane), ef[nt]);
      }
      float b2v[4];
#pragma unroll
      for (int nt = 0; nt < 4; ++nt) b2v[nt] = smem[O_EB2 + nt * 16 + ln];
#pragma unroll
      for (int j = 0; j < 8; ++j) {
        int m = g * 8 + j, e = tile * 16 + m;
        int rmj = escri[m * 12 + 10];
        bool val = (e < 462);
#pragma unroll
        for (int nt = 0; nt < 4; ++nt) {
          float v = silu_f(ef[nt][j] + b2v[nt]);
          if (val) atomicAdd(&SEG[rmj * 64 + nt * 16 + ln], v);
          float vo = __shfl_xor(v, 1);
          if (!(ln & 1)) scrbf[m * 36 + nt * 8 + (ln >> 1)] = pk_bf16(v, vo);
        }
      }
      __builtin_amdgcn_wave_barrier();

      // cm / cc heads: shared A = ef, silu + dot with w2, lane-butterfly reduce
      Frag a30 = afrag_pairs(scrbf, ln, 0, g);
      Frag a31 = afrag_pairs(scrbf, ln, 1, g);
      v8f hm[4] = {}, hc[4] = {};
#pragma unroll
      for (int nt = 0; nt < 4; ++nt) {
        hm[nt] = wmma_bf(a30, load_bfrag(CM1, 0, nt, lane), hm[nt]);
        hm[nt] = wmma_bf(a31, load_bfrag(CM1, 1, nt, lane), hm[nt]);
        hc[nt] = wmma_bf(a30, load_bfrag(CC1, 0, nt, lane), hc[nt]);
        hc[nt] = wmma_bf(a31, load_bfrag(CC1, 1, nt, lane), hc[nt]);
      }
      float cmb[4], ccb[4], cmw[4], ccw[4];
#pragma unroll
      for (int nt = 0; nt < 4; ++nt) {
        int n = nt * 16 + ln;
        cmb[nt] = smem[O_CMB + n];  ccb[nt] = smem[O_CCB + n];
        cmw[nt] = smem[O_CMW2 + n]; ccw[nt] = smem[O_CCW2 + n];
      }
#pragma unroll
      for (int j = 0; j < 8; ++j) {
        float pm = 0.f, pc = 0.f;
#pragma unroll
        for (int nt = 0; nt < 4; ++nt) {
          pm += silu_f(hm[nt][j] + cmb[nt]) * cmw[nt];
          pc += silu_f(hc[nt][j] + ccb[nt]) * ccw[nt];
        }
        for (int msk = 1; msk < 16; msk <<= 1) {
          pm += __shfl_xor(pm, msk, 16);
          pc += __shfl_xor(pc, msk, 16);
        }
        if (ln == 0) {
          escr[(g * 8 + j) * 12 + 8] = pm;
          escr[(g * 8 + j) * 12 + 9] = pc;
        }
      }
      __builtin_amdgcn_wave_barrier();

      // trans = coord_diff*cm + cc*cross -> accumulate into XA
      if (g == 0) {
        int e = tile * 16 + ln;
        if (e < 462) {
          int rm = escri[ln * 12 + 10];
          float cmv = escr[ln * 12 + 8], ccv = escr[ln * 12 + 9];
          atomicAdd(&XA[rm * 4 + 0], escr[ln * 12 + 2] * cmv + ccv * escr[ln * 12 + 5]);
          atomicAdd(&XA[rm * 4 + 1], escr[ln * 12 + 3] * cmv + ccv * escr[ln * 12 + 6]);
          atomicAdd(&XA[rm * 4 + 2], escr[ln * 12 + 4] * cmv + ccv * escr[ln * 12 + 7]);
        }
      }
    }
    __syncthreads();

    // bf16 mirror of completed segment sums for the node GEMM
    pack_rows(SEGBF, SEG, tid);
    __syncthreads();

    // ---- node phase: h += MLP([h, seg(ef)]); x += XA ----
    if (wave < 2) {
      int row = wave * 16 + ln;
      v8f acc[4] = {};
#pragma unroll
      for (int kt = 0; kt < 4; ++kt) {
        Frag a = (kt < 2) ? afrag_pairs(HBF, row, kt, g)
                          : afrag_pairs(SEGBF, row, kt - 2, g);
#pragma unroll
        for (int nt = 0; nt < 4; ++nt)
          acc[nt] = wmma_bf(a, load_bfrag(NW1, kt, nt, lane), acc[nt]);
      }
#pragma unroll
      for (int nt = 0; nt < 4; ++nt) {
        float bia = smem[O_NB1 + nt * 16 + ln];
#pragma unroll
        for (int j = 0; j < 8; ++j) {
          float v  = silu_f(acc[nt][j] + bia);
          float vo = __shfl_xor(v, 1);
          if (!(ln & 1))
            scrbf[(g * 8 + j) * 36 + nt * 8 + (ln >> 1)] = pk_bf16(v, vo);
        }
      }
      __builtin_amdgcn_wave_barrier();
      v8f o[4] = {};
#pragma unroll
      for (int kt = 0; kt < 2; ++kt) {
        Frag a = afrag_pairs(scrbf, ln, kt, g);
#pragma unroll
        for (int nt = 0; nt < 4; ++nt)
          o[nt] = wmma_bf(a, load_bfrag(NW2, kt, nt, lane), o[nt]);
      }
      __builtin_amdgcn_wave_barrier();
#pragma unroll
      for (int nt = 0; nt < 4; ++nt) {
        float bia = smem[O_NB2 + nt * 16 + ln];
#pragma unroll
        for (int j = 0; j < 8; ++j) {
          int rw = wave * 16 + g * 8 + j;
          if (rw < 22) H[rw * 64 + nt * 16 + ln] += o[nt][j] + bia;
        }
      }
    } else if (wave == 2) {
      for (int i = lane; i < 66; i += 32) {
        int p = i / 3, d = i - p * 3;
        X[p * 4 + d] += XA[p * 4 + d];
        XA[p * 4 + d] = 0.f;
      }
    }
    __syncthreads();
    // reset segment accumulator; ordered before next edge phase by the
    // post-stage __syncthreads of the next iteration
    for (int i = tid; i < 2048; i += 128) SEG[i] = 0.f;
  }

  // ---- output: vel = (x - x0) with per-graph mean removed ----
  if (tid < 3) {
    float s = 0.f;
    for (int p = 0; p < 22; ++p) s += X[p * 4 + tid] - X0[p * 4 + tid];
    smem[O_MEAN + tid] = s * (1.0f / 22.0f);
  }
  __syncthreads();
  for (int i = tid; i < 66; i += 128) {
    int p = i / 3, d = i - p * 3;
    out[b * 66 + i] = (X[p * 4 + d] - X0[p * 4 + d]) - smem[O_MEAN + d];
  }
}

extern "C" void kernel_launch(void* const* d_in, const int* in_sizes, int n_in,
                              void* d_out, int out_size, void* d_ws, size_t ws_size,
                              hipStream_t stream) {
  (void)in_sizes; (void)n_in; (void)out_size; (void)d_ws; (void)ws_size;
  const float* inx  = (const float*)d_in[0];
  const float* tarr = (const float*)d_in[1];
  const float* embw = (const float*)d_in[2];
  const float* embb = (const float*)d_in[3];
  const float* ew1  = (const float*)d_in[4];
  const float* eb1  = (const float*)d_in[5];
  const float* ew2  = (const float*)d_in[6];
  const float* eb2  = (const float*)d_in[7];
  const float* nw1  = (const float*)d_in[8];
  const float* nb1  = (const float*)d_in[9];
  const float* nw2  = (const float*)d_in[10];
  const float* nb2  = (const float*)d_in[11];
  const float* cmw1 = (const float*)d_in[12];
  const float* cmb1 = (const float*)d_in[13];
  const float* cmw2 = (const float*)d_in[14];
  const float* ccw1 = (const float*)d_in[15];
  const float* ccb1 = (const float*)d_in[16];
  const float* ccw2 = (const float*)d_in[17];
  // d_in[18]=row, d_in[19]=col: edge topology is computed analytically in-kernel.
  float* out = (float*)d_out;
  size_t smem_bytes = (size_t)SMEM_DW * sizeof(float);
  egnn_fused_kernel<<<dim3(1024), dim3(128), smem_bytes, stream>>>(
      inx, tarr, embw, embb, ew1, eb1, ew2, eb2, nw1, nb1, nw2, nb2,
      cmw1, cmb1, cmw2, ccw1, ccb1, ccw2, out);
}
